// MultiSimilarityLoss_37572373906132
// MI455X (gfx1250) — matrix operand order repaired and checked by reading, not compiled, not run on hardware
//
#include <hip/hip_runtime.h>
#include <hip/hip_bf16.h>
#include <math.h>

// ---- CDNA5 WMMA types ----
typedef __attribute__((ext_vector_type(16))) _Float16 v16h;
typedef __attribute__((ext_vector_type(8)))  _Float16 half8;
typedef __attribute__((ext_vector_type(8)))  float    v8f;

#define THRESH   0.5f
#define MARGIN   0.1f
#define SCALE_P  2.0f
#define SCALE_N  40.0f
#define ONE_MEPS 0.99999f   // 1 - 1e-5
#define FPOS_INF 3.0e38f

// ---------------------------------------------------------------------------
// K0: selection mask (argmax(score)==type_idx) + deterministic compaction via
//     block-wide inclusive scan. Single block of 1024 threads.
// ---------------------------------------------------------------------------
__global__ __launch_bounds__(1024)
void msl_select(const float* __restrict__ score, const int* __restrict__ labels,
                const int* __restrict__ type_idx_p, int B,
                int* __restrict__ idx_c, int* __restrict__ lab_c,
                int* __restrict__ S_out) {
    __shared__ int sd[1024];
    __shared__ int base;
    const int tid = threadIdx.x;
    if (tid == 0) base = 0;
    __syncthreads();
    const int ti = *type_idx_p;
    for (int c0 = 0; c0 < B; c0 += 1024) {
        const int i = c0 + tid;
        int sel = 0;
        if (i < B) {
            float s0 = score[i * 3 + 0];
            float s1 = score[i * 3 + 1];
            float s2 = score[i * 3 + 2];
            int am = 0; float bv = s0;
            if (s1 > bv) { bv = s1; am = 1; }
            if (s2 > bv) { bv = s2; am = 2; }
            sel = (am == ti) ? 1 : 0;
        }
        sd[tid] = sel;
        __syncthreads();
        for (int off = 1; off < 1024; off <<= 1) {
            int v = sd[tid];
            int add = (tid >= off) ? sd[tid - off] : 0;
            __syncthreads();
            sd[tid] = v + add;
            __syncthreads();
        }
        const int incl = sd[tid];
        const int rank = base + incl - sel;   // exclusive rank
        if (sel) { idx_c[rank] = i; lab_c[rank] = labels[i]; }
        __syncthreads();
        if (tid == 1023) base += incl;        // chunk total
        __syncthreads();
    }
    if (tid == 0) *S_out = base;
}

// ---------------------------------------------------------------------------
// K1: fp32 L2-normalize each selected row, store compact f16 matrix Fc[S][D].
// ---------------------------------------------------------------------------
__global__ __launch_bounds__(256)
void msl_normalize(const float* __restrict__ feats, const int* __restrict__ idx_c,
                   const int* __restrict__ S_p, int D,
                   _Float16* __restrict__ Fc) {
    const int s = blockIdx.x;
    if (s >= *S_p) return;
    const int r = idx_c[s];
    const float* src = feats + (size_t)r * D;
    __shared__ float red[256];
    float acc = 0.0f;
    for (int k = threadIdx.x; k < D; k += 256) { float v = src[k]; acc += v * v; }
    red[threadIdx.x] = acc;
    __syncthreads();
    for (int off = 128; off > 0; off >>= 1) {
        if (threadIdx.x < off) red[threadIdx.x] += red[threadIdx.x + off];
        __syncthreads();
    }
    const float inv = 1.0f / fmaxf(sqrtf(red[0]), 1e-12f);
    _Float16* dst = Fc + (size_t)s * D;
    for (int k = threadIdx.x; k < D; k += 256) dst[k] = (_Float16)(src[k] * inv);
}

#define SHUF16(lo, hi_) __builtin_shufflevector(lo, hi_, 0,1,2,3,4,5,6,7,8,9,10,11,12,13,14,15)

// ---------------------------------------------------------------------------
// 16x64 WMMA strip with software-pipelined (double-buffered) fragments:
// k-step N+1's loads are issued before k-step N's 4 WMMAs, so the matrix ops
// wait only on loads a full iteration old (partial s_wait_loadcnt, not 0).
// Per k-step: 2 A-b128 + 8 B-b128 feed 4 v_wmma (2.5 loads / wmma).
// ---------------------------------------------------------------------------
__device__ __forceinline__ void wmma_strip4(const _Float16* __restrict__ Fc, int D,
                                            int Mbase, const int col[4],
                                            int hi, int ln, v8f acc[4]) {
    const _Float16* arow = Fc + (size_t)(Mbase + ln) * D;   // A: M = lane&15
    const _Float16* brow[4];
#pragma unroll
    for (int t = 0; t < 4; ++t) brow[t] = Fc + (size_t)col[t] * D;

    // prologue: fragments for k0 = 0
    half8 ca0 = *(const half8*)(arow + hi * 8);
    half8 ca1 = *(const half8*)(arow + 16 + hi * 8);
    half8 cb0[4], cb1[4];
#pragma unroll
    for (int t = 0; t < 4; ++t) {
        cb0[t] = *(const half8*)(brow[t] + hi * 8);
        cb1[t] = *(const half8*)(brow[t] + 16 + hi * 8);
    }

#pragma unroll 2
    for (int k0 = 32; k0 < D; k0 += 32) {
        // prefetch next k-step into the second buffer
        half8 na0 = *(const half8*)(arow + k0 + hi * 8);
        half8 na1 = *(const half8*)(arow + k0 + 16 + hi * 8);
        half8 nb0[4], nb1[4];
#pragma unroll
        for (int t = 0; t < 4; ++t) {
            nb0[t] = *(const half8*)(brow[t] + k0 + hi * 8);
            nb1[t] = *(const half8*)(brow[t] + k0 + 16 + hi * 8);
        }
        // compute on the current buffer
        v16h A = SHUF16(ca0, ca1);
#pragma unroll
        for (int t = 0; t < 4; ++t) {
            v16h Bf = SHUF16(cb0[t], cb1[t]);
            acc[t] = __builtin_amdgcn_wmma_f32_16x16x32_f16(false, A, false, Bf,
                                                            (short)0, acc[t],
                                                            false, false);
        }
        ca0 = na0; ca1 = na1;
#pragma unroll
        for (int t = 0; t < 4; ++t) { cb0[t] = nb0[t]; cb1[t] = nb1[t]; }
    }
    // epilogue: last k-step
    v16h A = SHUF16(ca0, ca1);
#pragma unroll
    for (int t = 0; t < 4; ++t) {
        v16h Bf = SHUF16(cb0[t], cb1[t]);
        acc[t] = __builtin_amdgcn_wmma_f32_16x16x32_f16(false, A, false, Bf,
                                                        (short)0, acc[t],
                                                        false, false);
    }
}

// ---------------------------------------------------------------------------
// K2 (pass 1): per compact row -> count(same-label incl diag), min_pos, max_neg
//     Grid: ceil(B/16) blocks x 256 threads (8 waves). Block = one M tile.
//     Each wave sweeps 16x64 strips (4 N tiles) across all S columns.
// ---------------------------------------------------------------------------
__global__ __launch_bounds__(256)
void msl_pass1(const _Float16* __restrict__ Fc, const int* __restrict__ lab_c,
               const int* __restrict__ S_p, int D,
               float* __restrict__ min_pos, float* __restrict__ max_neg,
               int* __restrict__ cnt_g) {
    const int S = *S_p;
    const int Mbase = blockIdx.x * 16;
    if (Mbase >= S) return;
    const int tid = threadIdx.x;
    const int wave = tid >> 5, lane = tid & 31;
    const int hi = lane >> 4, ln = lane & 15;

    __shared__ int   rlab[16];
    __shared__ float smin[8][16];
    __shared__ float smax[8][16];
    __shared__ int   scnt[8][16];
    if (tid < 16) rlab[tid] = (Mbase + tid < S) ? lab_c[Mbase + tid] : -1;
    __syncthreads();

    float minp[8], maxn[8]; int cnt[8];
#pragma unroll
    for (int v = 0; v < 8; ++v) { minp[v] = FPOS_INF; maxn[v] = -FPOS_INF; cnt[v] = 0; }

    const int ntiles  = (S + 15) >> 4;
    const int ngroups = (ntiles + 3) >> 2;          // 64-col groups
    for (int g = wave; g < ngroups; g += 8) {
        int col[4]; int clab[4]; bool cvalid[4];
#pragma unroll
        for (int t = 0; t < 4; ++t) {
            col[t]    = g * 64 + t * 16 + ln;       // <= ngroups*64-1 <= 8191
            cvalid[t] = (col[t] < S);
            clab[t]   = cvalid[t] ? lab_c[col[t]] : 0x7fffffff;
        }
        v8f acc[4] = {{}, {}, {}, {}};
        wmma_strip4(Fc, D, Mbase, col, hi, ln, acc);

#pragma unroll
        for (int t = 0; t < 4; ++t) {
#pragma unroll
            for (int v = 0; v < 8; ++v) {
                const int m = v + 8 * hi;
                const int row = Mbase + m;
                const float s = acc[t][v];
                const bool leq = (rlab[m] == clab[t]);
                if (cvalid[t] & leq) {
                    cnt[v] += 1;
                    if ((row != col[t]) && (s < ONE_MEPS)) minp[v] = fminf(minp[v], s);
                } else if (cvalid[t]) {
                    maxn[v] = fmaxf(maxn[v], s);
                }
            }
        }
    }
    // reduce across the 16 lanes of each half-wave (rows v+8*hi)
#pragma unroll
    for (int v = 0; v < 8; ++v) {
        for (int off = 1; off < 16; off <<= 1) {
            minp[v] = fminf(minp[v], __shfl_xor(minp[v], off, 32));
            maxn[v] = fmaxf(maxn[v], __shfl_xor(maxn[v], off, 32));
            cnt[v] += __shfl_xor(cnt[v], off, 32);
        }
    }
    if (ln == 0) {
#pragma unroll
        for (int v = 0; v < 8; ++v) {
            smin[wave][hi * 8 + v] = minp[v];
            smax[wave][hi * 8 + v] = maxn[v];
            scnt[wave][hi * 8 + v] = cnt[v];
        }
    }
    __syncthreads();
    if (tid < 16 && (Mbase + tid) < S) {
        float mp = FPOS_INF, mn = -FPOS_INF; int ct = 0;
        for (int w = 0; w < 8; ++w) {
            mp = fminf(mp, smin[w][tid]);
            mn = fmaxf(mn, smax[w][tid]);
            ct += scnt[w][tid];
        }
        min_pos[Mbase + tid] = mp;
        max_neg[Mbase + tid] = mn;
        cnt_g[Mbase + tid] = ct;
    }
}

// ---------------------------------------------------------------------------
// K3 (pass 2): recompute sim strips, hard-pair exp-sums, per-row loss.
// ---------------------------------------------------------------------------
__global__ __launch_bounds__(256)
void msl_pass2(const _Float16* __restrict__ Fc, const int* __restrict__ lab_c,
               const int* __restrict__ S_p, int D, int B,
               const float* __restrict__ min_pos, const float* __restrict__ max_neg,
               const int* __restrict__ cnt_g, float* __restrict__ rowloss) {
    const int S = *S_p;
    const int Mbase = blockIdx.x * 16;
    const int tid = threadIdx.x;
    if (Mbase >= S) {                       // zero unused rows, keep output defined
        if (tid < 16 && (Mbase + tid) < B) rowloss[Mbase + tid] = 0.0f;
        return;
    }
    const int wave = tid >> 5, lane = tid & 31;
    const int hi = lane >> 4, ln = lane & 15;

    __shared__ int   rlab[16];
    __shared__ float rminp[16], rmaxn[16];
    __shared__ float sps[8][16], sns[8][16];
    __shared__ int   spa[8][16], sna[8][16];
    if (tid < 16) {
        const bool rv = (Mbase + tid) < S;
        rlab[tid]  = rv ? lab_c[Mbase + tid]   : -1;
        rminp[tid] = rv ? min_pos[Mbase + tid] : FPOS_INF;
        rmaxn[tid] = rv ? max_neg[Mbase + tid] : -FPOS_INF;
    }
    __syncthreads();

    float psum[8], nsum[8]; int pany[8], nany[8];
#pragma unroll
    for (int v = 0; v < 8; ++v) { psum[v] = 0.f; nsum[v] = 0.f; pany[v] = 0; nany[v] = 0; }

    const int ntiles  = (S + 15) >> 4;
    const int ngroups = (ntiles + 3) >> 2;
    for (int g = wave; g < ngroups; g += 8) {
        int col[4]; int clab[4]; bool cvalid[4];
#pragma unroll
        for (int t = 0; t < 4; ++t) {
            col[t]    = g * 64 + t * 16 + ln;
            cvalid[t] = (col[t] < S);
            clab[t]   = cvalid[t] ? lab_c[col[t]] : 0x7fffffff;
        }
        v8f acc[4] = {{}, {}, {}, {}};
        wmma_strip4(Fc, D, Mbase, col, hi, ln, acc);

#pragma unroll
        for (int t = 0; t < 4; ++t) {
#pragma unroll
            for (int v = 0; v < 8; ++v) {
                const int m = v + 8 * hi;
                const int row = Mbase + m;
                const float s = acc[t][v];
                const bool leq = (rlab[m] == clab[t]);
                if (cvalid[t] & leq) {
                    if ((row != col[t]) && (s < ONE_MEPS) && (s - MARGIN < rmaxn[m])) {
                        psum[v] += __expf(-SCALE_P * (s - THRESH));
                        pany[v] = 1;
                    }
                } else if (cvalid[t]) {
                    if (s + MARGIN > rminp[m]) {
                        nsum[v] += __expf(SCALE_N * (s - THRESH));
                        nany[v] = 1;
                    }
                }
            }
        }
    }
#pragma unroll
    for (int v = 0; v < 8; ++v) {
        for (int off = 1; off < 16; off <<= 1) {
            psum[v] += __shfl_xor(psum[v], off, 32);
            nsum[v] += __shfl_xor(nsum[v], off, 32);
            pany[v] |= __shfl_xor(pany[v], off, 32);
            nany[v] |= __shfl_xor(nany[v], off, 32);
        }
    }
    if (ln == 0) {
#pragma unroll
        for (int v = 0; v < 8; ++v) {
            sps[wave][hi * 8 + v] = psum[v];
            sns[wave][hi * 8 + v] = nsum[v];
            spa[wave][hi * 8 + v] = pany[v];
            sna[wave][hi * 8 + v] = nany[v];
        }
    }
    __syncthreads();
    if (tid < 16 && (Mbase + tid) < B) {
        const int row = Mbase + tid;
        float out = 0.0f;
        if (row < S) {
            float ps = 0.f, ns = 0.f; int pa = 0, na = 0;
            for (int w = 0; w < 8; ++w) {
                ps += sps[w][tid]; ns += sns[w][tid];
                pa |= spa[w][tid]; na |= sna[w][tid];
            }
            const bool haspos = min_pos[row] <  1.0e37f;
            const bool hasneg = max_neg[row] > -1.0e37f;
            const bool valid = (cnt_g[row] > 1) && haspos && hasneg && pa && na;
            if (valid) out = log1pf(ps) / SCALE_P + log1pf(ns) / SCALE_N;
        }
        rowloss[row] = out;
    }
}

// ---------------------------------------------------------------------------
// K4: deterministic final sum / B. Single block.
// ---------------------------------------------------------------------------
__global__ __launch_bounds__(256)
void msl_reduce(const float* __restrict__ rowloss, int B, float* __restrict__ out) {
    __shared__ float red[256];
    float acc = 0.0f;
    for (int i = threadIdx.x; i < B; i += 256) acc += rowloss[i];
    red[threadIdx.x] = acc;
    __syncthreads();
    for (int off = 128; off > 0; off >>= 1) {
        if (threadIdx.x < off) red[threadIdx.x] += red[threadIdx.x + off];
        __syncthreads();
    }
    if (threadIdx.x == 0) out[0] = red[0] / (float)B;
}

// ---------------------------------------------------------------------------
extern "C" void kernel_launch(void* const* d_in, const int* in_sizes, int n_in,
                              void* d_out, int out_size, void* d_ws, size_t ws_size,
                              hipStream_t stream) {
    const float* feats  = (const float*)d_in[0];
    const int*   labels = (const int*)d_in[1];
    const float* score  = (const float*)d_in[2];
    const int*   tidx   = (const int*)d_in[3];
    float* out = (float*)d_out;

    const int B = in_sizes[1];          // 8192
    const int D = in_sizes[0] / B;      // 1024 (multiple of 32)

    // workspace layout
    char* w = (char*)d_ws;
    int*   S_p     = (int*)w;                          // 256B reserved
    int*   idx_c   = (int*)(w + 256);
    int*   lab_c   = idx_c + B;
    float* min_pos = (float*)(lab_c + B);
    float* max_neg = min_pos + B;
    int*   cnt_g   = (int*)(max_neg + B);
    float* rowloss = (float*)(cnt_g + B);
    _Float16* Fc   = (_Float16*)(w + 256 + (size_t)B * 4 * 6); // 16B aligned

    const int mtiles = (B + 15) / 16;

    msl_select   <<<1, 1024, 0, stream>>>(score, labels, tidx, B, idx_c, lab_c, S_p);
    msl_normalize<<<B, 256, 0, stream>>>(feats, idx_c, S_p, D, Fc);
    msl_pass1    <<<mtiles, 256, 0, stream>>>(Fc, lab_c, S_p, D, min_pos, max_neg, cnt_g);
    msl_pass2    <<<mtiles, 256, 0, stream>>>(Fc, lab_c, S_p, D, B, min_pos, max_neg,
                                              cnt_g, rowloss);
    msl_reduce   <<<1, 256, 0, stream>>>(rowloss, B, out);
}
